// SessionGraph_87522843558042
// MI455X (gfx1250) — matrix operand order repaired and checked by reading, not compiled, not run on hardware
//
#include <hip/hip_runtime.h>
#include <hip/hip_bf16.h>
#include <math.h>

#define BB   512
#define LL   20
#define NSS  16
#define EPSS 32
#define HH   128
#define VV   100000
#define NN   (BB*NSS)     /* 8192 graph nodes  */
#define EE   (BB*EPSS)    /* 16384 edges       */
#define VM1  (VV-1)       /* 99999 score cols  */

typedef __attribute__((ext_vector_type(16))) _Float16 v16h;
typedef __attribute__((ext_vector_type(8)))  _Float16 h8;
typedef __attribute__((ext_vector_type(8)))  float    v8f;

__device__ __forceinline__ float sigm(float x) { return 1.0f / (1.0f + expf(-x)); }

// ---------------------------------------------------------------- K1: x = emb[items], agg = 0
__global__ void k_gather_zero(const int* __restrict__ items, const float* __restrict__ emb,
                              float* __restrict__ x, float* __restrict__ agg) {
    int i = blockIdx.x * blockDim.x + threadIdx.x;
    if (i >= NN * HH) return;
    int n = i >> 7, h = i & (HH - 1);
    x[i]   = emb[(size_t)items[n] * HH + h];
    agg[i] = 0.0f;
}

// ---------------------------------------------------------------- K2: m = x @ Wg   (Wg row-major [K][J])
__global__ void k_mgemm(const float* __restrict__ x, const float* __restrict__ Wg,
                        float* __restrict__ m) {
    __shared__ float xr[HH];
    int n = blockIdx.x, j = threadIdx.x;
    xr[j] = x[(size_t)n * HH + j];
    __syncthreads();
    float acc = 0.0f;
#pragma unroll 8
    for (int k = 0; k < HH; ++k) acc += xr[k] * Wg[k * HH + j];
    m[(size_t)n * HH + j] = acc;
}

// ---------------------------------------------------------------- K3: agg[dst] += m[src]
__global__ void k_scatter(const int* __restrict__ ei, const float* __restrict__ m,
                          float* __restrict__ agg) {
    int e = blockIdx.x, h = threadIdx.x;
    int src = ei[e], dst = ei[EE + e];
    atomicAdd(&agg[(size_t)dst * HH + h], m[(size_t)src * HH + h]);
}

// ---------------------------------------------------------------- K4: fused GRU cell per node
__global__ void k_gru(const float* __restrict__ x, const float* __restrict__ agg,
                      const float* __restrict__ W_ih, const float* __restrict__ W_hh,
                      const float* __restrict__ b_ih, const float* __restrict__ b_hh,
                      float* __restrict__ xn) {
    __shared__ float ar[HH], xr[HH];
    int n = blockIdx.x, h = threadIdx.x;
    ar[h] = agg[(size_t)n * HH + h];
    xr[h] = x[(size_t)n * HH + h];
    __syncthreads();
    float gi_r = b_ih[h], gi_z = b_ih[HH + h], gi_n = b_ih[2 * HH + h];
    float gh_r = b_hh[h], gh_z = b_hh[HH + h], gh_n = b_hh[2 * HH + h];
    const float* wr = W_ih + (size_t)h * HH;
    const float* wz = W_ih + (size_t)(HH + h) * HH;
    const float* wn = W_ih + (size_t)(2 * HH + h) * HH;
    const float* ur = W_hh + (size_t)h * HH;
    const float* uz = W_hh + (size_t)(HH + h) * HH;
    const float* un = W_hh + (size_t)(2 * HH + h) * HH;
#pragma unroll 4
    for (int k = 0; k < HH; ++k) {
        float a = ar[k], xv = xr[k];
        gi_r += wr[k] * a;  gi_z += wz[k] * a;  gi_n += wn[k] * a;
        gh_r += ur[k] * xv; gh_z += uz[k] * xv; gh_n += un[k] * xv;
    }
    float r  = sigm(gi_r + gh_r);
    float z  = sigm(gi_z + gh_z);
    float nn = tanhf(gi_n + r * gh_n);
    xn[(size_t)n * HH + h] = (1.0f - z) * nn + z * xr[h];
}

// ---------------------------------------------------------------- K5: ragged gather -> hidden[B,L,H]
__global__ void k_hidden(const int* __restrict__ items, const int* __restrict__ seq,
                         const float* __restrict__ xn, const float* __restrict__ emb,
                         float* __restrict__ hidden) {
    int bl = blockIdx.x;
    int b = bl / LL, h = threadIdx.x;
    int s = seq[bl];
    float v;
    if (s == 0) {
        v = 0.0f;
    } else {
        int idx = -1;
#pragma unroll
        for (int ns = 0; ns < NSS; ++ns)
            if (idx < 0 && items[b * NSS + ns] == s) idx = ns;
        v = (idx >= 0) ? xn[((size_t)b * NSS + idx) * HH + h]
                       : emb[(size_t)s * HH + h];
    }
    hidden[(size_t)bl * HH + h] = v;
}

// ---------------------------------------------------------------- K6a: ht (last item) and q1 = W1 ht + b1
__global__ void k_ht_q1(const int* __restrict__ mask, const float* __restrict__ hidden,
                        const float* __restrict__ W1, const float* __restrict__ b1,
                        float* __restrict__ ht, float* __restrict__ q1) {
    __shared__ float hr[HH];
    int b = blockIdx.x, h = threadIdx.x;
    int len = 0;
#pragma unroll
    for (int l = 0; l < LL; ++l) len += mask[b * LL + l];
    int pos = len - 1;
    float v = hidden[((size_t)b * LL + pos) * HH + h];
    ht[(size_t)b * HH + h] = v;
    hr[h] = v;
    __syncthreads();
    float acc = b1[h];
    const float* w = W1 + (size_t)h * HH;
#pragma unroll 8
    for (int k = 0; k < HH; ++k) acc += w[k] * hr[k];
    q1[(size_t)b * HH + h] = acc;
}

// ---------------------------------------------------------------- K6b: alpha[b,l] = sum_h sig(q1+q2)*w3
__global__ void k_alpha(const float* __restrict__ hidden, const float* __restrict__ W2,
                        const float* __restrict__ b2, const float* __restrict__ q1,
                        const float* __restrict__ w3, float* __restrict__ alpha) {
    __shared__ float hr[HH], red[HH];
    int bl = blockIdx.x;
    int b = bl / LL, h = threadIdx.x;
    hr[h] = hidden[(size_t)bl * HH + h];
    __syncthreads();
    float acc = b2[h];
    const float* w = W2 + (size_t)h * HH;
#pragma unroll 8
    for (int k = 0; k < HH; ++k) acc += w[k] * hr[k];
    red[h] = sigm(q1[(size_t)b * HH + h] + acc) * w3[h];
    __syncthreads();
    for (int s = HH / 2; s > 0; s >>= 1) {
        if (h < s) red[h] += red[h + s];
        __syncthreads();
    }
    if (h == 0) alpha[bl] = red[0];
}

// ---------------------------------------------------------------- K6c: pool + hybrid transform + layernorm -> f16
__global__ void k_pool_ln(const float* __restrict__ hidden, const float* __restrict__ alpha,
                          const int* __restrict__ mask, const float* __restrict__ ht,
                          const float* __restrict__ Wt, const float* __restrict__ bt,
                          const float* __restrict__ gamma, const float* __restrict__ beta,
                          _Float16* __restrict__ a16) {
    __shared__ float ar[HH], hr[HH], red[HH];
    int b = blockIdx.x, h = threadIdx.x;
    float acc = 0.0f;
#pragma unroll
    for (int l = 0; l < LL; ++l) {
        float mf = (float)mask[b * LL + l];
        acc += alpha[b * LL + l] * hidden[((size_t)b * LL + l) * HH + h] * mf;
    }
    ar[h] = acc;
    hr[h] = ht[(size_t)b * HH + h];
    __syncthreads();
    float a2 = bt[h];
    const float* w = Wt + (size_t)h * 2 * HH;
#pragma unroll 8
    for (int k = 0; k < HH; ++k) a2 += w[k] * ar[k] + w[HH + k] * hr[k];
    // layernorm over H within the block
    red[h] = a2; __syncthreads();
    for (int s = HH / 2; s > 0; s >>= 1) { if (h < s) red[h] += red[h + s]; __syncthreads(); }
    float mu = red[0] * (1.0f / HH);
    __syncthreads();
    float d = a2 - mu;
    red[h] = d * d; __syncthreads();
    for (int s = HH / 2; s > 0; s >>= 1) { if (h < s) red[h] += red[h + s]; __syncthreads(); }
    float var = red[0] * (1.0f / HH);
    float y = d * rsqrtf(var + 1e-5f) * gamma[h] + beta[h];
    a16[(size_t)b * HH + h] = (_Float16)y;
}

// ---------------------------------------------------------------- K7: scores = A[512,128] @ emb[1:].T  (WMMA)
// One wave per 16-column vocab tile; B fragments held in registers and reused
// for all 32 row tiles so the 51 MB embedding streams from HBM exactly once.
__global__ __launch_bounds__(32) void k_scores(const _Float16* __restrict__ A,
                                               const float* __restrict__ emb,
                                               float* __restrict__ out) {
    const int lane  = threadIdx.x;          // 0..31, one wave per block
    const int n0    = blockIdx.x * 16;      // vocab tile base
    const int col   = n0 + (lane & 15);
    const bool colOK = (col < VM1);
    const int erow  = colOK ? (1 + col) : VM1;   // clamp (valid row), stores guarded
    const int klow  = (lane < 16) ? 0 : 8;       // B: lanes 0-15 -> K {0-7,16-23}, 16-31 -> {8-15,24-31}
    const int koff  = (lane < 16) ? 0 : 8;       // A: lanes 0-15 -> K {0-7,16-23}, 16-31 -> {8-15,24-31}
    const int moff  = (lane < 16) ? 0 : 8;       // C/D: lanes 16-31 hold rows m+8
    const int mlane = lane & 15;

    // Load + convert B fragments for full K = 128 (4 chunks of 32)
    v16h bfrag[4];
#pragma unroll
    for (int it = 0; it < 4; ++it) {
        const float* src = emb + (size_t)erow * HH + it * 32 + klow;
        v16h bv;
#pragma unroll
        for (int e = 0; e < 16; ++e) bv[e] = (_Float16)src[e + (e & 8)];
        bfrag[it] = bv;
    }

    for (int mt = 0; mt < BB / 16; ++mt) {
        const int m0 = mt * 16;
        v8f c;
#pragma unroll
        for (int g = 0; g < 8; ++g) c[g] = 0.0f;

#pragma unroll
        for (int it = 0; it < 4; ++it) {
            const _Float16* arow = A + (size_t)(m0 + mlane) * HH + it * 32 + koff;
            h8 lo = *(const h8*)(arow);        // K koff..koff+7
            h8 hi = *(const h8*)(arow + 16);   // K koff+16..koff+23
            v16h av;
#pragma unroll
            for (int e = 0; e < 8; ++e) { av[e] = lo[e]; av[8 + e] = hi[e]; }
            c = __builtin_amdgcn_wmma_f32_16x16x32_f16(
                    /*neg_a=*/false, av, /*neg_b=*/false, bfrag[it],
                    /*c_mod=*/(short)0, c, /*reuse_a=*/false, /*reuse_b=*/false);
        }
        if (colOK) {
#pragma unroll
            for (int g = 0; g < 8; ++g)
                out[(size_t)(m0 + g + moff) * VM1 + col] = c[g];
        }
    }
}

// ----------------------------------------------------------------
extern "C" void kernel_launch(void* const* d_in, const int* in_sizes, int n_in,
                              void* d_out, int out_size, void* d_ws, size_t ws_size,
                              hipStream_t stream) {
    (void)in_sizes; (void)n_in; (void)out_size; (void)ws_size;
    const int*   items = (const int*)d_in[0];
    const int*   ei    = (const int*)d_in[1];   // [2, E]
    const int*   seq   = (const int*)d_in[2];
    const int*   mask  = (const int*)d_in[3];
    const float* emb   = (const float*)d_in[4];
    const float* Wg    = (const float*)d_in[5]; // [1,H,H]
    const float* W_ih  = (const float*)d_in[6];
    const float* W_hh  = (const float*)d_in[7];
    const float* b_ih  = (const float*)d_in[8];
    const float* b_hh  = (const float*)d_in[9];
    const float* W1    = (const float*)d_in[10];
    const float* b1    = (const float*)d_in[11];
    const float* W2    = (const float*)d_in[12];
    const float* b2    = (const float*)d_in[13];
    const float* w3    = (const float*)d_in[14];
    const float* Wt    = (const float*)d_in[15];
    const float* bt    = (const float*)d_in[16];
    const float* gamma = (const float*)d_in[17];
    const float* beta  = (const float*)d_in[18];
    float* out = (float*)d_out;

    // workspace layout (floats)
    float* ws     = (float*)d_ws;
    float* x      = ws;                       // N*H
    float* m      = x      + (size_t)NN * HH; // N*H
    float* agg    = m      + (size_t)NN * HH; // N*H
    float* xn     = agg    + (size_t)NN * HH; // N*H
    float* hidden = xn     + (size_t)NN * HH; // B*L*H
    float* ht     = hidden + (size_t)BB * LL * HH;
    float* q1     = ht     + (size_t)BB * HH;
    float* alpha  = q1     + (size_t)BB * HH; // B*L
    _Float16* a16 = (_Float16*)(alpha + (size_t)BB * LL); // B*H f16 (16B aligned)

    k_gather_zero<<<(NN * HH + 255) / 256, 256, 0, stream>>>(items, emb, x, agg);
    k_mgemm     <<<NN,      HH, 0, stream>>>(x, Wg, m);
    k_scatter   <<<EE,      HH, 0, stream>>>(ei, m, agg);
    k_gru       <<<NN,      HH, 0, stream>>>(x, agg, W_ih, W_hh, b_ih, b_hh, xn);
    k_hidden    <<<BB * LL, HH, 0, stream>>>(items, seq, xn, emb, hidden);
    k_ht_q1     <<<BB,      HH, 0, stream>>>(mask, hidden, W1, b1, ht, q1);
    k_alpha     <<<BB * LL, HH, 0, stream>>>(hidden, W2, b2, q1, w3, alpha);
    k_pool_ln   <<<BB,      HH, 0, stream>>>(hidden, alpha, mask, ht, Wt, bt, gamma, beta, a16);
    k_scores    <<<(VM1 + 15) / 16, 32, 0, stream>>>(a16, emb, out);
}
